// Mlp_13099650253522
// MI455X (gfx1250) — matrix operand order repaired and checked by reading, not compiled, not run on hardware
//
#include <hip/hip_runtime.h>
#include <math.h>

typedef _Float16 half_t;
typedef __attribute__((ext_vector_type(4)))  _Float16 v4h;
typedef __attribute__((ext_vector_type(8)))  _Float16 v8h;
typedef __attribute__((ext_vector_type(16))) _Float16 v16h;
typedef __attribute__((ext_vector_type(8)))  float    v8f;

#define LDS_PITCH 40   // 32 halves + 8 pad -> 80B rows, 16B aligned, conflict-free b128

// Low 32 bits of a flat shared-memory address = LDS byte offset (aperture layout).
__device__ __forceinline__ unsigned lds_off(const void* p) {
    return (unsigned)(uintptr_t)p;
}

// CDNA5 async DMA: memory -> LDS, tracked by ASYNCcnt (no VGPR staging).
__device__ __forceinline__ void async_load_b128(unsigned ldsaddr, const void* gptr) {
    asm volatile("global_load_async_to_lds_b128 %0, %1, off"
                 :: "v"(ldsaddr), "v"((unsigned long long)(uintptr_t)gptr)
                 : "memory");
}

// ---------------------------------------------------------------- reductions
__global__ __launch_bounds__(256) void absum_kernel(const float* __restrict__ w,
                                                    int n, float* __restrict__ out) {
    __shared__ float red[256];
    float s = 0.0f;
    for (int i = blockIdx.x * 256 + threadIdx.x; i < n; i += gridDim.x * 256)
        s += fabsf(w[i]);
    red[threadIdx.x] = s;
    __syncthreads();
    for (int off = 128; off > 0; off >>= 1) {
        if ((int)threadIdx.x < off) red[threadIdx.x] += red[threadIdx.x + off];
        __syncthreads();
    }
    if (threadIdx.x == 0) atomicAdd(out, red[0]);
}

// ------------------------------------------------- ternarize weights -> f16
__global__ __launch_bounds__(256) void quant_kernel(const float* __restrict__ w,
                                                    half_t* __restrict__ wq,
                                                    const float* __restrict__ gsum,
                                                    float invN, int n) {
    float gamma = gsum[0] * invN + 1e-5f;
    float inv = 1.0f / gamma;
    int i = blockIdx.x * 256 + threadIdx.x;
    if (i < n) {
        float q = rintf(w[i] * inv);           // RNE like jnp.round
        q = fminf(1.0f, fmaxf(-1.0f, q));      // {-1,0,1}; gamma folded in epilogue
        wq[i] = (half_t)q;
    }
}

// ---------------------------------------------------------- x: f32 -> f16
__global__ __launch_bounds__(256) void cvt_kernel(const float* __restrict__ in,
                                                  half_t* __restrict__ out, int n4) {
    int i = blockIdx.x * 256 + threadIdx.x;
    if (i < n4) {
        float4 v = ((const float4*)in)[i];
        v4h o = {(half_t)v.x, (half_t)v.y, (half_t)v.z, (half_t)v.w};
        ((v4h*)out)[i] = o;
    }
}

// --------------------------------------------------------------- WMMA GEMM
// C[M,N] = A[M,K] * Bq[N,K]^T, ternary Bq rescaled by gamma in epilogue.
// Block tile 128x256x32, 8 waves in 2(M)x4(N), wave tile 64x64 = 4x4 WMMAs.
// Double-buffered LDS fed by global_load_async_to_lds_b128 (ASYNCcnt).
template <int KDIM, bool GELU_F16OUT>
__global__ __launch_bounds__(256) void gemm_ternary(const half_t* __restrict__ A,
                                                    const half_t* __restrict__ Bq,
                                                    const float* __restrict__ bias,
                                                    const float* __restrict__ gsum,
                                                    float invN,
                                                    void* __restrict__ outp,
                                                    int M, int N) {
    constexpr int BM = 128, BN = 256, BK = 32;
    __shared__ half_t lds_a[2][BM * LDS_PITCH];
    __shared__ half_t lds_b[2][BN * LDS_PITCH];

    const int m0 = blockIdx.x * BM;
    const int n0 = blockIdx.y * BN;
    const int t = threadIdx.x;
    const int wave = t >> 5;
    const int lane = t & 31;
    const int wm = wave >> 2;          // 0..1 -> 64-row strip
    const int wn = wave & 3;           // 0..3 -> 64-col strip
    const int lr = lane & 15;
    const int lhi = lane >> 4;

    // Issue one 128x32 A tile + 256x32 B tile as async DMA (6 x b128 per thread).
    auto stage = [&](int k0, int buf) {
#pragma unroll
        for (int i = 0; i < 2; ++i) {               // A: 512 x 16B chunks
            int id = t + 256 * i;
            int r = id >> 2, cc = id & 3;
            int gm = m0 + r; gm = gm < M ? gm : M - 1;   // clamp: no OOB, rows unused
            async_load_b128(lds_off(&lds_a[buf][r * LDS_PITCH + cc * 8]),
                            A + (size_t)gm * KDIM + k0 + cc * 8);
        }
#pragma unroll
        for (int i = 0; i < 4; ++i) {               // B: 1024 x 16B chunks
            int id = t + 256 * i;
            int r = id >> 2, cc = id & 3;
            async_load_b128(lds_off(&lds_b[buf][r * LDS_PITCH + cc * 8]),
                            Bq + (size_t)(n0 + r) * KDIM + k0 + cc * 8);
        }
    };

    v8f acc[4][4] = {};

    stage(0, 0);
    constexpr int NSTEPS = KDIM / BK;
    for (int it = 0; it < NSTEPS; ++it) {
        const int buf = it & 1;
        if (it + 1 < NSTEPS) {
            stage((it + 1) * BK, buf ^ 1);          // prefetch next stage (6 in flight)
            asm volatile("s_wait_asynccnt 0x6" ::: "memory");
        } else {
            asm volatile("s_wait_asynccnt 0x0" ::: "memory");
        }
        __syncthreads();                            // stage `it` visible to all waves

        // ---- fragments per ISA 16-bit A(16x32)/B(32x16) lane maps
        v16h af[4];
#pragma unroll
        for (int mt = 0; mt < 4; ++mt) {
            int row = wm * 64 + mt * 16 + lr;
            v8h a0 = *(const v8h*)&lds_a[buf][row * LDS_PITCH + 8 * lhi];
            v8h a1 = *(const v8h*)&lds_a[buf][row * LDS_PITCH + 16 + 8 * lhi];
            af[mt] = __builtin_shufflevector(a0, a1, 0, 1, 2, 3, 4, 5, 6, 7,
                                             8, 9, 10, 11, 12, 13, 14, 15);
        }
        v16h bf[4];
#pragma unroll
        for (int nt = 0; nt < 4; ++nt) {
            int row = wn * 64 + nt * 16 + lr;
            v8h b0 = *(const v8h*)&lds_b[buf][row * LDS_PITCH + 8 * lhi];
            v8h b1 = *(const v8h*)&lds_b[buf][row * LDS_PITCH + 16 + 8 * lhi];
            bf[nt] = __builtin_shufflevector(b0, b1, 0, 1, 2, 3, 4, 5, 6, 7,
                                             8, 9, 10, 11, 12, 13, 14, 15);
        }
        // ---- 16 WMMAs per wave per k-step
#pragma unroll
        for (int mt = 0; mt < 4; ++mt)
#pragma unroll
            for (int nt = 0; nt < 4; ++nt)
                acc[mt][nt] = __builtin_amdgcn_wmma_f32_16x16x32_f16(
                    false, af[mt], false, bf[nt], (short)0, acc[mt][nt], false, false);

        __syncthreads();                            // release buffer for next DMA
    }

    // ---- epilogue: y = gamma*acc + bias (+ exact GELU, f16 out for layer 1)
    const float gamma = gsum[0] * invN + 1e-5f;
#pragma unroll
    for (int mt = 0; mt < 4; ++mt) {
#pragma unroll
        for (int nt = 0; nt < 4; ++nt) {
            int col = n0 + wn * 64 + nt * 16 + lr;
            float bv = bias[col];
#pragma unroll
            for (int vi = 0; vi < 8; ++vi) {
                int row = m0 + wm * 64 + mt * 16 + vi + lhi * 8;
                if (row < M) {
                    float v = gamma * acc[mt][nt][vi] + bv;
                    if constexpr (GELU_F16OUT) {
                        v = 0.5f * v * (1.0f + erff(v * 0.70710678118654752f));
                        ((half_t*)outp)[(size_t)row * N + col] = (half_t)v;
                    } else {
                        ((float*)outp)[(size_t)row * N + col] = v;
                    }
                }
            }
        }
    }
}

// ------------------------------------------------------------------- launch
extern "C" void kernel_launch(void* const* d_in, const int* in_sizes, int n_in,
                              void* d_out, int out_size, void* d_ws, size_t ws_size,
                              hipStream_t stream) {
    const float* x  = (const float*)d_in[0];
    const float* w1 = (const float*)d_in[1];
    const float* b1 = (const float*)d_in[2];
    const float* w2 = (const float*)d_in[3];
    const float* b2 = (const float*)d_in[4];

    const int H = in_sizes[2];            // 3072
    const int D = in_sizes[4];            // 768
    const int M = in_sizes[0] / D;        // 12608
    const int nW = H * D;

    char* ws = (char*)d_ws;
    float* gsum  = (float*)ws;                      // [0]=|w1| sum, [1]=|w2| sum
    half_t* xh   = (half_t*)(ws + 256);
    half_t* w1q  = xh + (size_t)M * D;
    half_t* w2q  = w1q + (size_t)nW;
    half_t* hbuf = w2q + (size_t)nW;

    hipMemsetAsync(gsum, 0, 256, stream);

    absum_kernel<<<1024, 256, 0, stream>>>(w1, nW, gsum + 0);
    absum_kernel<<<1024, 256, 0, stream>>>(w2, nW, gsum + 1);

    const float invNW = 1.0f / (float)nW;
    quant_kernel<<<(nW + 255) / 256, 256, 0, stream>>>(w1, w1q, gsum + 0, invNW, nW);
    quant_kernel<<<(nW + 255) / 256, 256, 0, stream>>>(w2, w2q, gsum + 1, invNW, nW);

    const int nx4 = (M * D) / 4;
    cvt_kernel<<<(nx4 + 255) / 256, 256, 0, stream>>>(x, xh, nx4);

    dim3 grid1((M + 127) / 128, H / 256);
    gemm_ternary<768, true><<<grid1, 256, 0, stream>>>(
        xh, w1q, b1, gsum + 0, invNW, (void*)hbuf, M, H);

    dim3 grid2((M + 127) / 128, D / 256);
    gemm_ternary<3072, false><<<grid2, 256, 0, stream>>>(
        hbuf, w2q, b2, gsum + 1, invNW, d_out, M, D);
}
